// PCNet_82197084110891
// MI455X (gfx1250) — matrix operand order repaired and checked by reading, not compiled, not run on hardware
//
#include <hip/hip_runtime.h>
#include <hip/hip_bf16.h>

typedef __attribute__((ext_vector_type(16))) _Float16 v16h;
typedef __attribute__((ext_vector_type(4)))  _Float16 v4h;
typedef __attribute__((ext_vector_type(8)))  float    v8f;
typedef __attribute__((ext_vector_type(4)))  float    v4f;   // native vec for nontemporal builtin

#define IN_F   500
#define H_F    64
#define NCLS   16
#define KCH    96      // K-chunk for LDS staging (multiple of 32; 500-480 is 4-aligned)
#define NCHUNK 6       // 6*96 = 576 >= 500 (tail zero-padded)

// ---------------------------------------------------------------------------
// Fused MLP: y = relu(x @ W1 + b1) @ W2 + b2     (f16 WMMA, f32 accumulate)
// One 256-thread block = 8 waves, each wave owns a 16-row output slab.
// LDS: W1 chunk transposed [64][96] (shared), W2 transposed [16][64] (shared),
//      per-wave x tile [16][96] and h tile [16][64].  Total ~54 KB.
// ---------------------------------------------------------------------------
__global__ __launch_bounds__(256) void pcnet_mlp(
    const float* __restrict__ x,  const float* __restrict__ W1,
    const float* __restrict__ b1, const float* __restrict__ W2,
    const float* __restrict__ b2, float* __restrict__ y, int n)
{
  __shared__ _Float16 sW1T[H_F * KCH];     // [col][k]
  __shared__ _Float16 sW2T[NCLS * H_F];    // [col][k]
  __shared__ _Float16 sX[8][16 * KCH];     // per-wave [row][k]
  __shared__ _Float16 sH[8][16 * H_F];     // per-wave [row][k]

  const int tid  = threadIdx.x;
  const int lane = tid & 31;
  const int wave = tid >> 5;
  const int ln   = lane & 15;
  const int hs   = (lane < 16) ? 0 : 1;       // half-wave select
  const int rowBase = blockIdx.x * 128 + wave * 16;

  // Stage W2 transposed once (contiguous global reads).
  for (int i = tid; i < NCLS * H_F; i += 256) {
    int c = i & 15, k = i >> 4;
    sW2T[c * H_F + k] = (_Float16)W2[k * NCLS + c];
  }

  // Accumulators init with b1 (C-layout: lane holds column ln, rows i/i+8).
  v8f acc[4];
  for (int ct = 0; ct < 4; ++ct) {
    float bv = b1[ct * 16 + ln];
    for (int i = 0; i < 8; ++i) acc[ct][i] = bv;
  }

  _Float16* sXw = sX[wave];
  _Float16* sHw = sH[wave];

  for (int kc = 0; kc < NCHUNK; ++kc) {
    const int k0 = kc * KCH;
    __syncthreads();
    // W1 chunk: coalesced float4 global reads, transposed scatter into LDS.
    for (int i = tid; i < KCH * (H_F / 4); i += 256) {
      int kk = i >> 4, q = i & 15;
      int k = k0 + kk;
      v4f v = (v4f)(0.f);
      if (k < IN_F) v = *reinterpret_cast<const v4f*>(W1 + (size_t)k * H_F + q * 4);
      int c = q * 4;
      sW1T[(c + 0) * KCH + kk] = (_Float16)v.x;
      sW1T[(c + 1) * KCH + kk] = (_Float16)v.y;
      sW1T[(c + 2) * KCH + kk] = (_Float16)v.z;
      sW1T[(c + 3) * KCH + kk] = (_Float16)v.w;
    }
    // Per-wave x tile [16][96]: non-temporal float4 loads (x is streamed once),
    // packed v4h (ds_store_b64) into LDS.
    for (int i = lane; i < 16 * (KCH / 4); i += 32) {
      int r = i / (KCH / 4);
      int q = i - r * (KCH / 4);
      int k = k0 + q * 4;
      int row = rowBase + r;
      v4f v = (v4f)(0.f);
      if (row < n && (k + 3) < IN_F)
        v = __builtin_nontemporal_load(
            reinterpret_cast<const v4f*>(x + (size_t)row * IN_F + k));
      v4h h4;
      h4[0] = (_Float16)v.x; h4[1] = (_Float16)v.y;
      h4[2] = (_Float16)v.z; h4[3] = (_Float16)v.w;
      *reinterpret_cast<v4h*>(sXw + r * KCH + q * 4) = h4;
    }
    // Prefetch next x chunk into cache while this one is consumed.
    if (kc + 1 < NCHUNK) {
      int row = rowBase + ln;
      int k = k0 + KCH + hs * 48;
      if (row < n && k < IN_F)
        __builtin_prefetch(x + (size_t)row * IN_F + k, 0, 0);
    }
    __syncthreads();

    for (int s = 0; s < 3; ++s) {           // 3 K-steps of 32 per chunk
      const int ks = s * 32;
      // A fragment (16-bit A 16x32 layout: lane<16 K={0..7,16..23}, else +8)
      v16h a;
      {
        const _Float16* ap = sXw + ln * KCH + ks + hs * 8;
        for (int j = 0; j < 8; ++j) a[j]     = ap[j];
        for (int j = 0; j < 8; ++j) a[8 + j] = ap[16 + j];
      }
      // Load all four B fragments first so the WMMAs can issue back-to-back.
      v16h bf[4];
      for (int ct = 0; ct < 4; ++ct) {
        const _Float16* bp = sW1T + (ct * 16 + ln) * KCH + ks + hs * 8;
        for (int j = 0; j < 8; ++j) bf[ct][j]     = bp[j];
        for (int j = 0; j < 8; ++j) bf[ct][8 + j] = bp[16 + j];
      }
      for (int ct = 0; ct < 4; ++ct)
        acc[ct] = __builtin_amdgcn_wmma_f32_16x16x32_f16(
            false, a, false, bf[ct], (short)0, acc[ct], false, false);
    }
  }

  // relu + re-layout h through per-wave LDS (C-layout -> A-layout).
  for (int ct = 0; ct < 4; ++ct)
    for (int i = 0; i < 8; ++i) {
      int M = i + hs * 8;
      sHw[M * H_F + ct * 16 + ln] = (_Float16)fmaxf(acc[ct][i], 0.f);
    }

  v8f acc2;
  {
    float bv = b2[ln];
    for (int i = 0; i < 8; ++i) acc2[i] = bv;
  }
  for (int s = 0; s < 2; ++s) {             // K = 64 = 2 steps of 32
    const int ks = s * 32;
    v16h a;
    const _Float16* ap = sHw + ln * H_F + ks + hs * 8;
    for (int j = 0; j < 8; ++j) a[j]     = ap[j];
    for (int j = 0; j < 8; ++j) a[8 + j] = ap[16 + j];
    v16h b;
    const _Float16* bp = sW2T + ln * H_F + ks + hs * 8;
    for (int j = 0; j < 8; ++j) b[j]     = bp[j];
    for (int j = 0; j < 8; ++j) b[8 + j] = bp[16 + j];
    acc2 = __builtin_amdgcn_wmma_f32_16x16x32_f16(
        false, a, false, b, (short)0, acc2, false, false);
  }
  for (int i = 0; i < 8; ++i) {
    int row = rowBase + i + hs * 8;
    if (row < n) y[row * NCLS + ln] = acc2[i];
  }
}

// ---------------------------------------------------------------------------
// Degree computation: deg = 1 (self loop) + scatter count; then deg^-0.5.
// ---------------------------------------------------------------------------
__global__ void pcnet_deg_init(float* d1, float* d2, int n) {
  int i = blockIdx.x * blockDim.x + threadIdx.x;
  if (i < n) { d1[i] = 1.f; d2[i] = 1.f; }
}

__global__ void pcnet_deg_count(const int* __restrict__ er,
                                const int* __restrict__ ec,
                                float* d1, float* d2, int e) {
  int i = blockIdx.x * blockDim.x + threadIdx.x;
  if (i < e) {
    unsafeAtomicAdd(&d1[er[i]], 1.f);
    unsafeAtomicAdd(&d2[ec[i]], 1.f);
  }
}

__global__ void pcnet_rdeg(float* d1, float* d2, int n) {
  int i = blockIdx.x * blockDim.x + threadIdx.x;
  if (i < n) { d1[i] = rsqrtf(d1[i]); d2[i] = rsqrtf(d2[i]); }
}

// ---------------------------------------------------------------------------
// wj[j] = (1/j!) * sum_{i=0..9} temp[1+i] * C_j(i+1),   Charlier recurrence
//   C0=1, C1=x-1, Cn=(x-n)C_{n-1}-(n-1)C_{n-2}   (lam = 1)
// Uniform over the wave -> compiles to SALU; negligible cost.
// ---------------------------------------------------------------------------
__device__ __forceinline__ float pcnet_wj(const float* __restrict__ temp, int j) {
  float fact = 1.f;
  for (int m = 2; m <= j; ++m) fact *= (float)m;
  float s = 0.f;
  for (int i = 0; i < 10; ++i) {
    float xx = (float)(i + 1);
    float c0 = 1.f, c1 = xx - 1.f, cn;
    if (j == 0) cn = c0;
    else if (j == 1) cn = c1;
    else {
      cn = c1;
      for (int m = 2; m <= j; ++m) { cn = (xx - (float)m) * c1 - (float)(m - 1) * c0; c0 = c1; c1 = cn; }
    }
    s += temp[1 + i] * cn;
  }
  return s / fact;
}

// ---------------------------------------------------------------------------
// Fused self-term init + polynomial accumulate (float4-vectorized).
//   nxt[i] = (r1[i]*r2[i] - 1) * cur[i]
//   mode==0 : out = (temp[0] + wj[0]) * cur        (cur == y, T0 term)
//   mode>0  : out += wj[mode] * cur                (accumulate previous hop)
// ---------------------------------------------------------------------------
__global__ void pcnet_selfinit_acc(v4f* __restrict__ nxt, const v4f* __restrict__ cur,
                                   const float* __restrict__ r1, const float* __restrict__ r2,
                                   v4f* __restrict__ out, const float* __restrict__ temp,
                                   int mode, int n4) {
  int t = blockIdx.x * blockDim.x + threadIdx.x;
  if (t >= n4) return;
  int i = t >> 2;
  float s = r1[i] * r2[i] - 1.f;
  v4f v = cur[t];
  v4f o;
  float w = pcnet_wj(temp, mode);
  if (mode == 0) {
    o = (temp[0] + w) * v;
  } else {
    o = out[t] + w * v;
  }
  out[t] = o;
  nxt[t] = s * v;
}

// Final accumulate: out += wj[4] * cur (float4-vectorized).
__global__ void pcnet_final_acc(v4f* __restrict__ out, const v4f* __restrict__ cur,
                                const float* __restrict__ temp, int n4) {
  int t = blockIdx.x * blockDim.x + threadIdx.x;
  if (t >= n4) return;
  float w = pcnet_wj(temp, 4);
  out[t] = out[t] + w * cur[t];
}

// ---------------------------------------------------------------------------
// Edge scatter: nxt[col] += r1[row]*r2[col] * cur[row]
// 4 lanes per edge (one float4 quad each) -> coalesced 64B row gathers,
// native global_atomic_add_f32 (no return).
// ---------------------------------------------------------------------------
__global__ void pcnet_scatter(const int* __restrict__ er, const int* __restrict__ ec,
                              const float* __restrict__ r1, const float* __restrict__ r2,
                              const float* __restrict__ cur, float* __restrict__ nxt,
                              int e) {
  int t = blockIdx.x * blockDim.x + threadIdx.x;
  if (t < 4 * e) {
    int ei = t >> 2, q = t & 3;
    int r = er[ei], c = ec[ei];
    float w = r1[r] * r2[c];
    const v4f v = *reinterpret_cast<const v4f*>(cur + (size_t)r * 16 + q * 4);
    float* dst = nxt + (size_t)c * 16 + q * 4;
    unsafeAtomicAdd(dst + 0, w * v.x);
    unsafeAtomicAdd(dst + 1, w * v.y);
    unsafeAtomicAdd(dst + 2, w * v.z);
    unsafeAtomicAdd(dst + 3, w * v.w);
  }
}

// ---------------------------------------------------------------------------
extern "C" void kernel_launch(void* const* d_in, const int* in_sizes, int n_in,
                              void* d_out, int out_size, void* d_ws, size_t ws_size,
                              hipStream_t stream) {
  const float* x    = (const float*)d_in[0];
  const int*   ei   = (const int*)d_in[1];
  const float* W1   = (const float*)d_in[2];
  const float* b1   = (const float*)d_in[3];
  const float* W2   = (const float*)d_in[4];
  const float* b2   = (const float*)d_in[5];
  const float* temp = (const float*)d_in[6];
  float* out = (float*)d_out;

  const int n = in_sizes[0] / IN_F;
  const int e = in_sizes[1] / 2;
  const int* er = ei;        // edge_index[0]
  const int* ec = ei + e;    // edge_index[1]

  float* ws   = (float*)d_ws;
  float* y    = ws;                          // [n*16]
  float* bufA = y + (size_t)n * 16;          // [n*16]
  float* bufB = bufA + (size_t)n * 16;       // [n*16]
  float* d1   = bufB + (size_t)n * 16;       // [n]
  float* d2   = d1 + n;                      // [n]

  const int n4 = n * 4;                      // float4 quads per [n*16] buffer
  dim3 blk(256);

  pcnet_mlp<<<dim3((n + 127) / 128), blk, 0, stream>>>(x, W1, b1, W2, b2, y, n);
  pcnet_deg_init<<<dim3((n + 255) / 256), blk, 0, stream>>>(d1, d2, n);
  pcnet_deg_count<<<dim3((e + 255) / 256), blk, 0, stream>>>(er, ec, d1, d2, e);
  pcnet_rdeg<<<dim3((n + 255) / 256), blk, 0, stream>>>(d1, d2, n);

  // 4 hops: tmps[:5] = [y, c1, c2, c3, c4]
  const float* cur = y;
  for (int it = 1; it <= 4; ++it) {
    float* nxt = (it & 1) ? bufA : bufB;
    // mode it-1: accumulates wj[it-1]*cur into out (mode 0 also adds temp[0]*y)
    pcnet_selfinit_acc<<<dim3((n4 + 255) / 256), blk, 0, stream>>>(
        (v4f*)nxt, (const v4f*)cur, d1, d2, (v4f*)out, temp, it - 1, n4);
    pcnet_scatter<<<dim3((4 * e + 255) / 256), blk, 0, stream>>>(er, ec, d1, d2, cur, nxt, e);
    cur = nxt;
  }
  pcnet_final_acc<<<dim3((n4 + 255) / 256), blk, 0, stream>>>(
      (v4f*)out, (const v4f*)cur, temp, n4);
}
